// TransformerPAU_47751446397543
// MI455X (gfx1250) — matrix operand order repaired and checked by reading, not compiled, not run on hardware
//
#include <hip/hip_runtime.h>
#include <hip/hip_bf16.h>

// ---------------------------------------------------------------------------
// CDNA5 (gfx1250) wave32 WMMA implementation of the GNN + Transformer + conv
// head reference.  All GEMMs are bf16 x bf16 -> f32 accum (v_wmma 16x16x32),
// tiles staged to LDS with GLOBAL_LOAD_ASYNC_TO_LDS_B128 when available,
// double-buffered K loop, f32 residual stream + bf16 shadow copy.
// ---------------------------------------------------------------------------

typedef __bf16 bf16;
typedef __attribute__((ext_vector_type(16))) __bf16 v16bf;
typedef __attribute__((ext_vector_type(8)))  float  v8f;
typedef __attribute__((ext_vector_type(4)))  int    v4i;

#define WMMA_BF16(A_, B_, C_) \
  __builtin_amdgcn_wmma_f32_16x16x32_bf16(false, (A_), false, (B_), (short)0, (C_), false, false)

// Gate the gfx1250 async-to-LDS path on builtin availability (device pass).
#if defined(__has_builtin)
#if __has_builtin(__builtin_amdgcn_global_load_async_to_lds_b128)
#define HAVE_ASYNC_LDS 1
#endif
#endif
#ifndef HAVE_ASYNC_LDS
#define HAVE_ASYNC_LDS 0
#endif

typedef __attribute__((address_space(1))) v4i as1_v4i;   // global (prints as __device__)
typedef __attribute__((address_space(3))) v4i as3_v4i;   // LDS    (prints as __shared__)

// 16-byte global -> LDS copy (async on gfx1250, memcpy fallback elsewhere).
__device__ __forceinline__ void copy16_g2l(const void* g, void* l) {
#if HAVE_ASYNC_LDS
  __builtin_amdgcn_global_load_async_to_lds_b128((as1_v4i*)g, (as3_v4i*)l, 0, 0);
#else
  __builtin_memcpy(l, g, 16);
#endif
}
__device__ __forceinline__ void async_wait() {
#if HAVE_ASYNC_LDS
#if __has_builtin(__builtin_amdgcn_s_wait_asynccnt)
  __builtin_amdgcn_s_wait_asynccnt(0);
#else
  asm volatile("s_wait_asynccnt 0x0" ::: "memory");
#endif
#endif
}

// A-matrix 16x32 bf16 lane layout used here (contiguous k; B uses the same):
// lane = half*16 + m ; element e of the 16-element vector maps to
// K = 16*half + e
__device__ __forceinline__ int a_kofs(int e, int half) {
  return (half << 4) | e;
}

__device__ __forceinline__ void store_out(float* p, float v) { *p = v; }
__device__ __forceinline__ void store_out(bf16*  p, float v) { *p = (bf16)v; }

__device__ __forceinline__ float gelu_exact(float x) {
  return 0.5f * x * (1.0f + erff(x * 0.70710678118654752f));
}

// ---------------------------------------------------------------------------
// f32 -> bf16 bulk conversion (weights once; X -> Xb shadow).
// ---------------------------------------------------------------------------
__global__ __launch_bounds__(256) void f32_to_bf16_kernel(const float* __restrict__ src,
                                                          bf16* __restrict__ dst, int n) {
  const int i = blockIdx.x * 256 + threadIdx.x;
  if (i < n) dst[i] = (bf16)src[i];
}

// ---------------------------------------------------------------------------
// Build X[B,256,256]: rows 0..63 = [h_prev | bond_feat], rows 64..255 = 0.
// ---------------------------------------------------------------------------
__global__ __launch_bounds__(256) void init_x_kernel(const float* __restrict__ h_prev,
                                                     float* __restrict__ X) {
  const int row = blockIdx.x;            // b*256 + s
  const int d   = threadIdx.x;           // 0..255
  const int b = row >> 8, s = row & 255;
  float v = 0.f;
  if (s < 64 && d < 128) v = h_prev[((size_t)b * 64 + s) * 128 + d];
  X[(size_t)row * 256 + d] = v;
}

// Bond encoder + weighted segment aggregation.  One block (128 threads) per
// graph; thread t owns embedding channel t.
__global__ __launch_bounds__(128) void bond_kernel(const int*   __restrict__ edge_src,
                                                   const int*   __restrict__ edge_attr,
                                                   const float* __restrict__ bond_emb,
                                                   float* __restrict__ X) {
  __shared__ int   la[128], at0[128], at1[128], at2[128];
  __shared__ float wle[128];
  __shared__ int   cnt[64], starts[64];

  const int b = blockIdx.x, t = threadIdx.x;
  if (t < 64) cnt[t] = 0;
  __syncthreads();

  const int e = t;                              // 128 edges, 128 threads
  const int src = edge_src[b * 128 + e];        // global atom id
  const int lat = src - b * 64;
  la[e]  = lat;
  at0[e] = edge_attr[(b * 128 + e) * 3 + 0];
  at1[e] = edge_attr[(b * 128 + e) * 3 + 1];
  at2[e] = edge_attr[(b * 128 + e) * 3 + 2];
  atomicAdd(&cnt[lat], 1);
  __syncthreads();

  if (t < 64) { int s = 0; for (int j = 0; j < t; ++j) s += cnt[j]; starts[t] = s; }
  __syncthreads();

  {
    const int k = cnt[la[e]];
    const int pos = e - starts[la[e]];
    wle[e] = (k == 1) ? 1.f
           : ((pos == 0) ? exp2f((float)(1 - k)) : exp2f((float)(pos - k)));
  }
  __syncthreads();

  const float* be0 = bond_emb;
  const float* be1 = bond_emb + 16 * 128;
  const float* be2 = bond_emb + 32 * 128;

  float acc = 0.f;
  int cur = la[0];
  for (int i = 0; i < 128; ++i) {
    const int a = la[i];
    if (a != cur) {
      X[((size_t)b * 256 + cur) * 256 + 128 + t] = acc;
      acc = 0.f; cur = a;
    }
    const float emb = be0[at0[i] * 128 + t] + be1[at1[i] * 128 + t] + be2[at2[i] * 128 + t];
    acc = fmaf(wle[i], emb, acc);
  }
  X[((size_t)b * 256 + cur) * 256 + 128 + t] = acc;
}

// ---------------------------------------------------------------------------
// bf16 WMMA GEMM:  C[M,N] = epi(A[M,K] * W[K,N] + bias[N])
// Block tile 128x128, K-step 32, 256 threads = 8 waves (4x2 accum tiles each).
// Double-buffered LDS, async global->LDS staging.  EPI: 0=bias, 1=bias+GELU.
// ---------------------------------------------------------------------------
template <typename TC, int EPI>
__global__ __launch_bounds__(256) void gemm_kernel(const bf16* __restrict__ A,
                                                   const bf16* __restrict__ W,
                                                   const float* __restrict__ bias,
                                                   TC* __restrict__ C,
                                                   int M, int N, int K) {
  __shared__ __align__(16) bf16 As[2][128][40];   // [m][k], 80B row stride
  __shared__ __align__(16) bf16 Bs[2][32][136];   // [k][n], 272B row stride

  const int tid  = threadIdx.x;
  const int wave = tid >> 5, lane = tid & 31;
  const int half = lane >> 4, lidx = lane & 15;

  const int bm = blockIdx.y << 7;
  const int bn = blockIdx.x << 7;
  const int wm = (wave & 1) << 6;    // 0 / 64
  const int wn = (wave >> 1) << 5;   // 0 / 32 / 64 / 96

  v8f acc[4][2];
  for (int m_ = 0; m_ < 4; ++m_)
    for (int n_ = 0; n_ < 2; ++n_)
      for (int r = 0; r < 8; ++r) acc[m_][n_][r] = 0.f;

  // stage one 128x32 A tile + 32x128 B tile (16B chunks, 4 per thread)
  auto stage = [&](int buf, int k0) {
    for (int i = tid; i < 512; i += 256) {                 // A: 4 chunks/row
      const int r = i >> 2, c = (i & 3) << 3;
      copy16_g2l(&A[(size_t)(bm + r) * K + (k0 + c)], &As[buf][r][c]);
    }
    for (int i = tid; i < 512; i += 256) {                 // B: 16 chunks/row
      const int r = i >> 4, c = (i & 15) << 3;
      copy16_g2l(&W[(size_t)(k0 + r) * N + (bn + c)], &Bs[buf][r][c]);
    }
  };

  stage(0, 0);
  int cur = 0;
  for (int k0 = 0; k0 < K; k0 += 32) {
    async_wait();
    __syncthreads();
    if (k0 + 32 < K) stage(cur ^ 1, k0 + 32);

    v16bf afr[4], bfr[2];
#pragma unroll
    for (int m_ = 0; m_ < 4; ++m_) {
      const int row = wm + m_ * 16 + lidx;
#pragma unroll
      for (int e = 0; e < 16; ++e) afr[m_][e] = As[cur][row][a_kofs(e, half)];
    }
#pragma unroll
    for (int n_ = 0; n_ < 2; ++n_) {
      const int col = wn + n_ * 16 + lidx;
#pragma unroll
      for (int e = 0; e < 16; ++e) bfr[n_][e] = Bs[cur][(half << 4) + e][col];
    }
#pragma unroll
    for (int m_ = 0; m_ < 4; ++m_)
#pragma unroll
      for (int n_ = 0; n_ < 2; ++n_)
        acc[m_][n_] = WMMA_BF16(afr[m_], bfr[n_], acc[m_][n_]);
    cur ^= 1;
  }

  for (int m_ = 0; m_ < 4; ++m_)
    for (int n_ = 0; n_ < 2; ++n_) {
      const int col = bn + wn + n_ * 16 + lidx;
      const float bv = bias[col];
      for (int r = 0; r < 8; ++r) {
        const int row = bm + wm + m_ * 16 + half * 8 + r;
        float v = acc[m_][n_][r] + bv;
        if (EPI == 1) v = gelu_exact(v);
        store_out(&C[(size_t)row * N + col], v);
      }
    }
}

// ---------------------------------------------------------------------------
// Attention: one block per (b, head, 32-row q block).  S=256, DH=128.
// scores = (q k^T) * 1/sqrt(128); +1 row-constant mask is softmax-invariant.
// ---------------------------------------------------------------------------
__global__ __launch_bounds__(256) void attn_kernel(const bf16* __restrict__ qkv,
                                                   bf16* __restrict__ out) {
  __shared__ __align__(16) bf16  qs[32][136];
  __shared__ __align__(16) bf16  kvs[64][136];
  __shared__ float sc[32][260];

  const int qb = blockIdx.x;   // 0..7
  const int h  = blockIdx.y;   // 0..1
  const int b  = blockIdx.z;   // 0..511

  const int tid  = threadIdx.x;
  const int wave = tid >> 5, lane = tid & 31;
  const int half = lane >> 4, lidx = lane & 15;

  const bf16* qp = qkv + (size_t)b * 256 * 768 + (size_t)h * 128;
  const bf16* kp = qp + 256;
  const bf16* vp = qp + 512;

  // stage q block (32x128 bf16 -> 512 x 16B chunks)
  for (int i = tid; i < 512; i += 256) {
    const int r = i >> 4, c = (i & 15) << 3;
    copy16_g2l(&qp[(size_t)(qb * 32 + r) * 768 + c], &qs[r][c]);
  }

  const float scale = 0.08838834764831845f;   // 1/sqrt(128)
  const int mi = wave & 1;                    // 2 M subtiles (32 q rows)

  // ---- scores: q (32x128) x k^T (128x256), 4 key tiles of 64 ----
  for (int kt = 0; kt < 4; ++kt) {
    for (int i = tid; i < 1024; i += 256) {   // 64x128 k tile
      const int r = i >> 4, c = (i & 15) << 3;
      copy16_g2l(&kp[(size_t)(kt * 64 + r) * 768 + c], &kvs[r][c]);
    }
    async_wait();
    __syncthreads();

    const int ni = wave >> 1;   // 0..3 within the 64-wide key tile
    v8f acc; for (int r = 0; r < 8; ++r) acc[r] = 0.f;
    for (int k0 = 0; k0 < 128; k0 += 32) {
      v16bf a, bb;
      const int row = mi * 16 + lidx;
#pragma unroll
      for (int e = 0; e < 16; ++e) a[e] = qs[row][k0 + a_kofs(e, half)];
      const int col = ni * 16 + lidx;     // key row within tile
#pragma unroll
      for (int e = 0; e < 16; ++e) bb[e] = kvs[col][k0 + (half << 4) + e];  // B = k^T
      acc = WMMA_BF16(a, bb, acc);
    }
    {
      const int col = kt * 64 + ni * 16 + lidx;
#pragma unroll
      for (int r = 0; r < 8; ++r)
        sc[mi * 16 + half * 8 + r][col] = acc[r] * scale;
    }
    __syncthreads();
  }

  // ---- softmax over 256 keys, one thread per q row ----
  if (tid < 32) {
    const int r = tid;
    float mx = -3.0e38f;
    for (int j = 0; j < 256; ++j) mx = fmaxf(mx, sc[r][j]);
    float s = 0.f;
    for (int j = 0; j < 256; ++j) { const float e = __expf(sc[r][j] - mx); sc[r][j] = e; s += e; }
    const float inv = 1.f / s;
    for (int j = 0; j < 256; ++j) sc[r][j] *= inv;
  }
  __syncthreads();

  // ---- O = P (32x256) x V (256x128) ----
  const int nset = wave >> 1;   // 0..3 -> dh cols nset*32
  v8f oacc[2];
  for (int nj = 0; nj < 2; ++nj)
    for (int r = 0; r < 8; ++r) oacc[nj][r] = 0.f;

  for (int vt = 0; vt < 4; ++vt) {
    for (int i = tid; i < 1024; i += 256) {   // 64x128 v tile
      const int r = i >> 4, c = (i & 15) << 3;
      copy16_g2l(&vp[(size_t)(vt * 64 + r) * 768 + c], &kvs[r][c]);
    }
    async_wait();
    __syncthreads();
    for (int k0 = 0; k0 < 64; k0 += 32) {
      v16bf a;
      const int row = mi * 16 + lidx;
#pragma unroll
      for (int e = 0; e < 16; ++e)
        a[e] = (bf16)sc[row][vt * 64 + k0 + a_kofs(e, half)];
#pragma unroll
      for (int nj = 0; nj < 2; ++nj) {
        v16bf bb;
        const int col = nset * 32 + nj * 16 + lidx;
#pragma unroll
        for (int e = 0; e < 16; ++e) bb[e] = kvs[k0 + (half << 4) + e][col];
        oacc[nj] = WMMA_BF16(a, bb, oacc[nj]);
      }
    }
    __syncthreads();
  }

  for (int nj = 0; nj < 2; ++nj) {
    const int col = nset * 32 + nj * 16 + lidx;
    for (int r = 0; r < 8; ++r) {
      const int row = mi * 16 + half * 8 + r;
      out[((size_t)b * 256 + qb * 32 + row) * 256 + h * 128 + col] = (bf16)oacc[nj][r];
    }
  }
}

// ---------------------------------------------------------------------------
// X = LayerNorm(X + P) row-wise (row length 256), one block per row.
// Also emits the bf16 shadow copy Xb used as GEMM input.
// ---------------------------------------------------------------------------
__global__ __launch_bounds__(256) void ln_residual_kernel(float* __restrict__ X,
                                                          bf16* __restrict__ Xb,
                                                          const float* __restrict__ P,
                                                          const float* __restrict__ g,
                                                          const float* __restrict__ bt) {
  __shared__ float red[256];
  const int row = blockIdx.x, t = threadIdx.x;
  const float v = X[(size_t)row * 256 + t] + P[(size_t)row * 256 + t];
  red[t] = v;
  __syncthreads();
  for (int s = 128; s > 0; s >>= 1) { if (t < s) red[t] += red[t + s]; __syncthreads(); }
  const float mean = red[0] * (1.f / 256.f);
  __syncthreads();
  const float d = v - mean;
  red[t] = d * d;
  __syncthreads();
  for (int s = 128; s > 0; s >>= 1) { if (t < s) red[t] += red[t + s]; __syncthreads(); }
  const float var = red[0] * (1.f / 256.f);
  const float y = d * rsqrtf(var + 1e-5f) * g[t] + bt[t];
  X[(size_t)row * 256 + t]  = y;
  Xb[(size_t)row * 256 + t] = (bf16)y;
}

// ---------------------------------------------------------------------------
// Conv head chain + final linear head.  One block per graph, thread s owns
// output column s; weights are thread-uniform -> scalar (SGPR) loads.
// ---------------------------------------------------------------------------
__global__ __launch_bounds__(256) void conv_head_kernel(const float* __restrict__ X,
    const float* __restrict__ cw0, const float* __restrict__ cb0,
    const float* __restrict__ cw1, const float* __restrict__ cb1,
    const float* __restrict__ cw2, const float* __restrict__ cb2,
    const float* __restrict__ cw3, const float* __restrict__ cb3,
    const float* __restrict__ head_w, const float* __restrict__ head_b,
    float* __restrict__ out) {
  const int b = blockIdx.x, s = threadIdx.x;
  const float* xb = X + (size_t)b * 256 * 256;

  float o0[128];
#pragma unroll
  for (int o = 0; o < 128; ++o) o0[o] = cb0[o];
  for (int c = 0; c < 256; ++c) {
    const float xv = xb[(size_t)c * 256 + s];
#pragma unroll
    for (int o = 0; o < 128; ++o) o0[o] = fmaf(cw0[o * 256 + c], xv, o0[o]);
  }

  float o1[64];
#pragma unroll
  for (int o = 0; o < 64; ++o) o1[o] = cb1[o];
  for (int c = 0; c < 128; ++c) {
    const float xv = o0[c];
#pragma unroll
    for (int o = 0; o < 64; ++o) o1[o] = fmaf(cw1[o * 128 + c], xv, o1[o]);
  }

  float o2[32];
#pragma unroll
  for (int o = 0; o < 32; ++o) o2[o] = cb2[o];
  for (int c = 0; c < 64; ++c) {
    const float xv = o1[c];
#pragma unroll
    for (int o = 0; o < 32; ++o) o2[o] = fmaf(cw2[o * 64 + c], xv, o2[o]);
  }

  float o3 = cb3[0];
#pragma unroll
  for (int c = 0; c < 32; ++c) o3 = fmaf(cw3[c], o2[c], o3);

  __shared__ float red[256];
  red[s] = o3 * head_w[s];
  __syncthreads();
  for (int st = 128; st > 0; st >>= 1) { if (s < st) red[s] += red[s + st]; __syncthreads(); }
  if (s == 0) out[b] = red[0] + head_b[0];
}

// ---------------------------------------------------------------------------
// Host-side orchestration
// ---------------------------------------------------------------------------
extern "C" void kernel_launch(void* const* d_in, const int* in_sizes, int n_in,
                              void* d_out, int out_size, void* d_ws, size_t ws_size,
                              hipStream_t stream) {
  (void)in_sizes; (void)n_in; (void)out_size; (void)ws_size;

  const float* h_prev    = (const float*)d_in[0];
  const int*   edge_src  = (const int*)d_in[1];
  const int*   edge_attr = (const int*)d_in[2];
  const float* bond_emb  = (const float*)d_in[3];
  const float* Wqkv = (const float*)d_in[4];
  const float* bqkv = (const float*)d_in[5];
  const float* Wo   = (const float*)d_in[6];
  const float* bo   = (const float*)d_in[7];
  const float* ln1g = (const float*)d_in[8];
  const float* ln1b = (const float*)d_in[9];
  const float* W1   = (const float*)d_in[10];
  const float* b1   = (const float*)d_in[11];
  const float* W2   = (const float*)d_in[12];
  const float* b2   = (const float*)d_in[13];
  const float* ln2g = (const float*)d_in[14];
  const float* ln2b = (const float*)d_in[15];
  const float* cw0 = (const float*)d_in[16];
  const float* cb0 = (const float*)d_in[17];
  const float* cw1 = (const float*)d_in[18];
  const float* cb1 = (const float*)d_in[19];
  const float* cw2 = (const float*)d_in[20];
  const float* cb2 = (const float*)d_in[21];
  const float* cw3 = (const float*)d_in[22];
  const float* cb3 = (const float*)d_in[23];
  const float* head_w = (const float*)d_in[24];
  const float* head_b = (const float*)d_in[25];
  float* out = (float*)d_out;

  const int Mrows = 512 * 256;                       // 131072
  const size_t XB  = (size_t)Mrows * 256 * 4;        // 134,217,728
  const size_t XbB = (size_t)Mrows * 256 * 2;        //  67,108,864
  const size_t UB  = (size_t)Mrows * 2048 * 2;       // 536,870,912

  char* ws = (char*)d_ws;
  float* X  = (float*)ws;                            // residual stream (f32)
  float* P  = (float*)(ws + XB);                     // GEMM f32 out / LN input
  bf16*  Xb = (bf16*)(ws + 2 * XB);                  // bf16 shadow of X
  char*  U  = ws + 2 * XB + XbB;                     // union scratch
  bf16* QKV = (bf16*)U;                              // 131072 x 768  (bf16)
  bf16* AO  = (bf16*)(U + (size_t)Mrows * 768 * 2);  // 131072 x 256  (bf16)
  bf16* F   = (bf16*)U;                              // 131072 x 2048 (bf16), aliases QKV/AO
  char* WB  = U + UB;                                // bf16 weight copies
  bf16* Wqkv_b = (bf16*)WB;                          // 4 x 256 x 768
  bf16* Wo_b   = (bf16*)(WB + 1572864);              // 4 x 256 x 256
  bf16* W1_b   = (bf16*)(WB + 1572864 + 524288);     // 4 x 256 x 2048
  bf16* W2_b   = (bf16*)(WB + 1572864 + 524288 + 4194304);  // 4 x 2048 x 256

  // one-time bf16 weight conversion (tiny: ~10.5 MB total)
  f32_to_bf16_kernel<<<(786432 + 255) / 256, 256, 0, stream>>>(Wqkv, Wqkv_b, 786432);
  f32_to_bf16_kernel<<<(262144 + 255) / 256, 256, 0, stream>>>(Wo, Wo_b, 262144);
  f32_to_bf16_kernel<<<(2097152 + 255) / 256, 256, 0, stream>>>(W1, W1_b, 2097152);
  f32_to_bf16_kernel<<<(2097152 + 255) / 256, 256, 0, stream>>>(W2, W2_b, 2097152);

  // Build X and its bf16 shadow
  init_x_kernel<<<Mrows, 256, 0, stream>>>(h_prev, X);
  bond_kernel<<<512, 128, 0, stream>>>(edge_src, edge_attr, bond_emb, X);
  f32_to_bf16_kernel<<<Mrows * 256 / 256, 256, 0, stream>>>(X, Xb, Mrows * 256);

  for (int l = 0; l < 4; ++l) {
    // QKV = Xb @ Wqkv[l] + bqkv[l]   (bf16 in, bf16 out)
    gemm_kernel<bf16, 0><<<dim3(6, 1024), 256, 0, stream>>>(
        Xb, Wqkv_b + (size_t)l * 256 * 768, bqkv + (size_t)l * 768, QKV,
        Mrows, 768, 256);

    // attention
    attn_kernel<<<dim3(8, 2, 512), 256, 0, stream>>>(QKV, AO);

    // P = AO @ Wo[l] + bo[l]         (bf16 in, f32 out)
    gemm_kernel<float, 0><<<dim3(2, 1024), 256, 0, stream>>>(
        AO, Wo_b + (size_t)l * 256 * 256, bo + (size_t)l * 256, P,
        Mrows, 256, 256);

    // X = LN(X + P); Xb = bf16(X)
    ln_residual_kernel<<<Mrows, 256, 0, stream>>>(X, Xb, P,
        ln1g + (size_t)l * 256, ln1b + (size_t)l * 256);

    // F = gelu(Xb @ W1[l] + b1[l])   (bf16 in, bf16 out)
    gemm_kernel<bf16, 1><<<dim3(16, 1024), 256, 0, stream>>>(
        Xb, W1_b + (size_t)l * 256 * 2048, b1 + (size_t)l * 2048, F,
        Mrows, 2048, 256);

    // P = F @ W2[l] + b2[l]          (bf16 in, f32 out)
    gemm_kernel<float, 0><<<dim3(2, 1024), 256, 0, stream>>>(
        F, W2_b + (size_t)l * 2048 * 256, b2 + (size_t)l * 256, P,
        Mrows, 256, 2048);

    // X = LN(X + P); Xb = bf16(X)
    ln_residual_kernel<<<Mrows, 256, 0, stream>>>(X, Xb, P,
        ln2g + (size_t)l * 256, ln2b + (size_t)l * 256);
  }

  // conv head chain + final linear
  conv_head_kernel<<<512, 256, 0, stream>>>(X, cw0, cb0, cw1, cb1, cw2, cb2,
                                            cw3, cb3, head_w, head_b, out);
}
